// ParallelEncoderLayer_37563783970985
// MI455X (gfx1250) — matrix operand order repaired and checked by reading, not compiled, not run on hardware
//
#include <hip/hip_runtime.h>
#include <hip/hip_bf16.h>

// ---------------------------------------------------------------------------
// ParallelEncoderLayer forward for MI455X (gfx1250, wave32, WMMA).
//
// Input flattening assumption:
//   d_in[0]=grid(B,G,D) f32, [1]=image_id int, [2]=enti2attr(B,R,2) int,
//   [3]=sub2obj2rela(B,R,3) int, [4]=sg_mask(2,B,R) bool(u8),
//   [5]=region(B,R,D) f32, [6]=boxes(B,R,4) f32,
//   [7..] = params flattened as a JAX pytree (dict keys sorted, None dropped):
//     7 Wa30 8 Wa49 9 Wg 10 Wr 11 ba30 12 ba49 13 bg 14 br
//     15-18 ff1{W1,W2,b1,b2} 19-22 ff2 23-26 ff3
//     27-33 fus0{Wa,We,Wf,Wo,ba,be,bo} 34-40 fus1
//     41-45 gcn{W_ea,W_rel,b_ea,b_rel,word_emb}
//     46 ln1_b 47 ln1_g 48 ln_b 49 ln_g
//     50-57 mha1{Wk,Wo,Wq,Wv,bk,bo,bq,bv} 58-65 mha2 66-73 mha3
// Output: image(B,79,D) | out_mask(B,79) | sg(B,R,D) | _sg_mask(B,R), all f32.
// ---------------------------------------------------------------------------

#define B_   32
#define G_   49
#define R_   30
#define D_   512
#define H_   8
#define DK_  64
#define DFF_ 2048
#define NPOS_ 101
#define BRG_ (B_*R_*G_)

typedef __attribute__((ext_vector_type(16))) _Float16 v16h;
typedef __attribute__((ext_vector_type(8)))  _Float16 v8h;
typedef __attribute__((ext_vector_type(8)))  float    v8f;
typedef __attribute__((ext_vector_type(4)))  float    v4f;
typedef long long ll;

// ================= weight pre-pass: (K,N) f32 -> (N,K) f16 ==================
__global__ __launch_bounds__(256)
void k_prep_w(const float* __restrict__ Wsrc, _Float16* __restrict__ Wdst, int K, int N)
{
    __shared__ _Float16 tile[32][33];
    const int bk = blockIdx.x * 32, bn = blockIdx.y * 32;
    const int tx = threadIdx.x & 31, ty = threadIdx.x >> 5;   // 32 x 8
    #pragma unroll
    for (int u = 0; u < 4; ++u) {
        int k = bk + ty + u * 8, n = bn + tx;
        tile[ty + u * 8][tx] = (_Float16)((k < K && n < N) ? Wsrc[(ll)k * N + n] : 0.f);
    }
    __syncthreads();
    #pragma unroll
    for (int u = 0; u < 4; ++u) {
        int n = bn + ty + u * 8, k = bk + tx;
        if (n < N && k < K) Wdst[(ll)n * K + k] = tile[tx][ty + u * 8];
    }
}

// ============================ WMMA GEMM =====================================
// Y[M,N] = act( A[M,K] @ W[K,N] + bias ), weights pre-transposed to (N,K) f16.
// AMODE 0: A element = A1[b*s0 + i*s1 + j*s2 + k], m -> (b,i,j) via (O1,O2)
// AMODE 1: A element = A1[(b*O1+i)*K + k] * A2[(b*O2+j)*K + k]
// ACT 0: none, 1: relu, 2: sigmoid(relu(x))
// Block tile 64x128x32, 8 waves x (32x32), 4 WMMA per wave per k-step.
// Double-buffered LDS; out-of-range rows/cols are address-clamped (their
// garbage results only land in guarded, never-stored output rows/cols).

template<int ACT>
__device__ __forceinline__ void store_tile(const v8f& c, float* __restrict__ Y,
                                           const float* __restrict__ bias,
                                           int row0, int col, int M, int N)
{
    float bv = bias ? bias[col] : 0.f;
    #pragma unroll
    for (int r = 0; r < 8; ++r) {
        int row = row0 + r;
        if (row < M) {
            float x = c[r] + bv;
            if (ACT == 1) x = fmaxf(x, 0.f);
            if (ACT == 2) { x = fmaxf(x, 0.f); x = 1.f / (1.f + __expf(-x)); }
            Y[(ll)row * N + col] = x;
        }
    }
}

template<int AMODE, int ACT>
__global__ __launch_bounds__(256)
void gemm_wmma(const float* __restrict__ A1, const float* __restrict__ A2,
               int O1, int O2, ll s0, ll s1, ll s2,
               const _Float16* __restrict__ WmT, const float* __restrict__ bias,
               float* __restrict__ Y, int M, int N, int K)
{
    __shared__ _Float16 As[2][64][40];     // [buf][m][k], rows 16B-aligned
    __shared__ _Float16 BsT[2][128][40];   // [buf][n][k], transposed weights

    const int t    = threadIdx.x;
    const int lane = t & 31;
    const int wave = t >> 5;
    const int bm   = blockIdx.x;
    const int bn   = blockIdx.y;

    const int wr = (wave >> 2) << 5;     // 0 / 32
    const int wc = (wave & 3) << 5;      // 0 / 32 / 64 / 96

    v8f c00 = {}, c01 = {}, c10 = {}, c11 = {};

    // ---- A tile loader: row = t/4 (0..63), k0 = (t%4)*8 ; address-clamped
    const int arow = t >> 2;
    const int ak0  = (t & 3) << 3;
    int m_g = bm * 64 + arow; if (m_g > M - 1) m_g = M - 1;
    ll ab1 = 0, ab2 = 0;
    {
        int oo  = O1 * O2;
        int b   = m_g / oo;
        int rem = m_g - b * oo;
        int i   = rem / O2;
        int j   = rem - i * O2;
        if (AMODE == 0) ab1 = (ll)b * s0 + (ll)i * s1 + (ll)j * s2 + ak0;
        else { ab1 = (ll)(b * O1 + i) * K + ak0; ab2 = (ll)(b * O2 + j) * K + ak0; }
    }
    // ---- B tile loader: n = t/2 (0..127), kk0 = (t%2)*16 ; address-clamped
    const int bln  = t >> 1;
    const int bk0  = (t & 1) << 4;
    int n_gb = bn * 128 + bln; if (n_gb > N - 1) n_gb = N - 1;
    const _Float16* wbase = WmT + (ll)n_gb * K + bk0;   // contiguous 16 halfs/tile

    const int half = lane >> 4;
    const int l16  = lane & 15;

    auto load_a = [&](ll koff) -> v8h {
        v4f f0 = *(const v4f*)(A1 + ab1 + koff);
        v4f f1 = *(const v4f*)(A1 + ab1 + koff + 4);
        if (AMODE == 1) {
            v4f g0 = *(const v4f*)(A2 + ab2 + koff);
            v4f g1 = *(const v4f*)(A2 + ab2 + koff + 4);
            f0 *= g0; f1 *= g1;
        }
        v8h h;
        #pragma unroll
        for (int u = 0; u < 4; ++u) { h[u] = (_Float16)f0[u]; h[4 + u] = (_Float16)f1[u]; }
        return h;
    };
    auto load_b = [&](ll koff, v8h& h0, v8h& h1) {
        h0 = *(const v8h*)(wbase + koff);
        h1 = *(const v8h*)(wbase + koff + 8);
    };
    auto stash = [&](int buf, const v8h& ha, const v8h& hb0, const v8h& hb1) {
        *(v8h*)&As[buf][arow][ak0]      = ha;
        *(v8h*)&BsT[buf][bln][bk0]      = hb0;
        *(v8h*)&BsT[buf][bln][bk0 + 8]  = hb1;
    };
    auto compute = [&](int buf) {
        const _Float16* pa0 = &As[buf][wr + l16][half << 3];
        const _Float16* pa1 = &As[buf][wr + 16 + l16][half << 3];
        v8h alo0 = *(const v8h*)pa0,  ahi0 = *(const v8h*)(pa0 + 16);
        v8h alo1 = *(const v8h*)pa1,  ahi1 = *(const v8h*)(pa1 + 16);
        v16h a0 = __builtin_shufflevector(alo0, ahi0, 0,1,2,3,4,5,6,7,8,9,10,11,12,13,14,15);
        v16h a1 = __builtin_shufflevector(alo1, ahi1, 0,1,2,3,4,5,6,7,8,9,10,11,12,13,14,15);
        const _Float16* pb0 = &BsT[buf][wc + l16][half << 3];
        const _Float16* pb1 = &BsT[buf][wc + 16 + l16][half << 3];
        v8h blo0 = *(const v8h*)pb0,  bhi0 = *(const v8h*)(pb0 + 16);
        v8h blo1 = *(const v8h*)pb1,  bhi1 = *(const v8h*)(pb1 + 16);
        v16h b0 = __builtin_shufflevector(blo0, bhi0, 0,1,2,3,4,5,6,7,8,9,10,11,12,13,14,15);
        v16h b1 = __builtin_shufflevector(blo1, bhi1, 0,1,2,3,4,5,6,7,8,9,10,11,12,13,14,15);
        c00 = __builtin_amdgcn_wmma_f32_16x16x32_f16(false, a0, false, b0, (short)0, c00, false, false);
        c01 = __builtin_amdgcn_wmma_f32_16x16x32_f16(false, a0, false, b1, (short)0, c01, false, false);
        c10 = __builtin_amdgcn_wmma_f32_16x16x32_f16(false, a1, false, b0, (short)0, c10, false, false);
        c11 = __builtin_amdgcn_wmma_f32_16x16x32_f16(false, a1, false, b1, (short)0, c11, false, false);
    };

    // ---- prologue: tile 0 into buffer 0
    const int nK = K >> 5;
    {
        v8h ha = load_a(0), hb0, hb1;
        load_b(0, hb0, hb1);
        stash(0, ha, hb0, hb1);
    }
    __syncthreads();

    // ---- main loop: prefetch k+1 while computing k; one barrier per step
    int cur = 0;
    for (int kb = 0; kb < nK - 1; ++kb) {
        ll koff = (ll)(kb + 1) << 5;
        v8h ha = load_a(koff);
        v8h hb0, hb1;
        load_b(koff, hb0, hb1);
        compute(cur);
        stash(cur ^ 1, ha, hb0, hb1);
        __syncthreads();
        cur ^= 1;
    }
    compute(cur);   // last tile

    const int rb = bm * 64 + wr + (half << 3);
    const int cb = bn * 128 + wc + l16;
    store_tile<ACT>(c00, Y, bias, rb,      cb,      M, N);
    store_tile<ACT>(c01, Y, bias, rb,      cb + 16, M, N);
    store_tile<ACT>(c10, Y, bias, rb + 16, cb,      M, N);
    store_tile<ACT>(c11, Y, bias, rb + 16, cb + 16, M, N);
}

// ============================ attention (per b,h) ===========================
__global__ __launch_bounds__(256)
void k_attn(const float* __restrict__ Q, const float* __restrict__ Kf,
            const float* __restrict__ V, float* __restrict__ O,
            const unsigned char* __restrict__ mask, int n)
{
    __shared__ float sq[G_][DK_], sk[G_][DK_], sv[G_][DK_], ss[G_][G_ + 1];
    const int b = blockIdx.x / H_;
    const int h = blockIdx.x % H_;
    const int t = threadIdx.x;

    for (int e = t; e < n * DK_; e += 256) {
        int i = e >> 6, d = e & 63;
        ll base = ((ll)(b * n + i)) * D_ + h * DK_ + d;
        sq[i][d] = Q[base]; sk[i][d] = Kf[base]; sv[i][d] = V[base];
    }
    __syncthreads();
    for (int e = t; e < n * n; e += 256) {
        int i = e / n, j = e - i * n;
        float s = 0.f;
        #pragma unroll 8
        for (int d = 0; d < DK_; ++d) s += sq[i][d] * sk[j][d];
        s *= 0.125f;
        if (mask && mask[b * n + j]) s = -1e9f;
        ss[i][j] = s;
    }
    __syncthreads();
    if (t < n) {
        float mx = -1e30f;
        for (int j = 0; j < n; ++j) mx = fmaxf(mx, ss[t][j]);
        float sum = 0.f;
        for (int j = 0; j < n; ++j) { float e = __expf(ss[t][j] - mx); ss[t][j] = e; sum += e; }
        float inv = 1.f / sum;
        for (int j = 0; j < n; ++j) ss[t][j] *= inv;
    }
    __syncthreads();
    for (int e = t; e < n * DK_; e += 256) {
        int i = e >> 6, d = e & 63;
        float o = 0.f;
        for (int j = 0; j < n; ++j) o += ss[i][j] * sv[j][d];
        O[((ll)(b * n + i)) * D_ + h * DK_ + d] = o;
    }
}

// ============================ residual + layernorm ==========================
__global__ __launch_bounds__(256)
void k_residual_ln(const float* __restrict__ X, const float* __restrict__ Res,
                   const float* __restrict__ gam, const float* __restrict__ bet,
                   float* __restrict__ Y, int M)
{
    int row  = blockIdx.x * 8 + (threadIdx.x >> 5);
    int lane = threadIdx.x & 31;
    if (row >= M) return;
    const float* x = X   + (ll)row * D_;
    const float* r = Res + (ll)row * D_;
    float vals[16];
    float s = 0.f;
    #pragma unroll
    for (int u = 0; u < 16; ++u) { float v = x[lane + u * 32] + r[lane + u * 32]; vals[u] = v; s += v; }
    #pragma unroll
    for (int off = 16; off; off >>= 1) s += __shfl_xor(s, off, 32);
    float mean = s * (1.f / D_);
    float vs = 0.f;
    #pragma unroll
    for (int u = 0; u < 16; ++u) { float d = vals[u] - mean; vs += d * d; }
    #pragma unroll
    for (int off = 16; off; off >>= 1) vs += __shfl_xor(vs, off, 32);
    float inv = rsqrtf(vs * (1.f / D_) + 1e-5f);
    #pragma unroll
    for (int u = 0; u < 16; ++u) {
        int c = lane + u * 32;
        float y = (vals[u] - mean) * inv;
        if (gam) y = y * gam[c] + bet[c];
        Y[(ll)row * D_ + c] = y;
    }
}

// ============================ small kernels =================================
__global__ void k_zero2(unsigned* p) { p[0] = 0u; p[1] = 0u; }

__global__ void k_sinusoid(float* tab) {
    int e = blockIdx.x * blockDim.x + threadIdx.x;
    if (e >= NPOS_ * D_) return;
    int p = e / D_, i = e - p * D_;
    float v = 0.f;
    if (p != 0) {
        float expo = 2.f * (float)(i / 2) / (float)D_;
        float ang  = (float)p / powf(10000.f, expo);
        v = (i & 1) ? cosf(ang) : sinf(ang);
    }
    tab[e] = v;
}

__global__ void k_gather_embed2(const int* __restrict__ e2a, const float* __restrict__ emb,
                                float* __restrict__ out) {
    ll e = (ll)blockIdx.x * blockDim.x + threadIdx.x;
    if (e >= (ll)B_ * R_ * 2 * D_) return;
    int col = (int)(e & 1023); ll m = e >> 10;
    int id = (col < D_) ? e2a[m * 2] : e2a[m * 2 + 1];
    out[e] = emb[(ll)id * D_ + (col & (D_ - 1))];
}

__global__ void k_gather_relobj(const int* __restrict__ s2o, const float* __restrict__ emb,
                                const float* __restrict__ sg, float* __restrict__ out) {
    ll e = (ll)blockIdx.x * blockDim.x + threadIdx.x;
    if (e >= (ll)B_ * R_ * D_) return;
    int k = (int)(e & (D_ - 1)); ll m = e >> 9;
    int b = (int)(m / R_);
    int rel = s2o[m * 3 + 2], obj = s2o[m * 3 + 1];
    out[e] = emb[(ll)rel * D_ + k] + sg[((ll)(b * R_ + obj)) * D_ + k];
}

__global__ void k_scatter_add(const int* __restrict__ s2o, const float* __restrict__ msg,
                              float* __restrict__ sg) {
    ll e = (ll)blockIdx.x * blockDim.x + threadIdx.x;
    if (e >= (ll)B_ * R_ * D_) return;
    int k = (int)(e & (D_ - 1)); ll m = e >> 9;
    int b = (int)(m / R_);
    int dst = s2o[m * 3 + 0];
    atomicAdd(&sg[((ll)(b * R_ + dst)) * D_ + k], msg[e]);
}

__global__ void k_geom1(const float* __restrict__ boxes, float* __restrict__ ca,
                        float* __restrict__ cl, float* __restrict__ cp, unsigned* mx) {
    int m = blockIdx.x * blockDim.x + threadIdx.x;
    if (m >= BRG_) return;
    int g = m % G_; int br = m / G_;
    const float* bx = boxes + (ll)br * 4;
    float rx1 = bx[0] * (1.f/800.f), ry1 = bx[1] * (1.f/600.f);
    float rx2 = bx[2] * (1.f/800.f), ry2 = bx[3] * (1.f/600.f);
    int ii = g / 7, jj = g % 7;
    float gx1 = jj * (1.f/7.f), gy1 = ii * (1.f/7.f);
    float gx2 = gx1 + 1.f/7.f,  gy2 = gy1 + 1.f/7.f;
    float cwx = fabsf(0.5f*(rx1+rx2) - 0.5f*(gx1+gx2));
    float cwy = fabsf(0.5f*(ry1+ry2) - 0.5f*(gy1+gy2));
    float clen = cwx * cwy;
    float mxx = fmaxf(rx1, gx1), mxy = fmaxf(ry1, gy1);
    float mnx = fminf(rx2, gx2), mny = fminf(ry2, gy2);
    bool noov = (mxx >= mnx) || (mxy >= mny);
    float wx = fabsf(mnx - mxx), wy = fabsf(mny - mxy);
    float cov = noov ? 0.f : wx * wy;
    float rarea = (rx2 - rx1) * (ry2 - ry1);
    ca[m] = cov; cl[m] = clen; cp[m] = cov / (rarea + 1e-9f);
    atomicMax(mx + 0, __float_as_uint(cov));
    atomicMax(mx + 1, __float_as_uint(clen));
}

__global__ void k_geom2(const float* __restrict__ ca, const float* __restrict__ cl,
                        float* __restrict__ cp, const unsigned* __restrict__ mx,
                        const unsigned char* __restrict__ mask1,
                        float* __restrict__ extra, int* __restrict__ idx) {
    int m = blockIdx.x * blockDim.x + threadIdx.x;
    if (m >= BRG_) return;
    int br = m / G_;
    float camax = __uint_as_float(mx[0]);
    float clmax = __uint_as_float(mx[1]);
    float e = cp[m] + ca[m] / camax - (cl[m] * cl[m]) / clmax;
    extra[m] = e;
    int id = (int)(fmaxf(e, 0.f) * (100.f / 3.f));
    if (id > NPOS_ - 1) id = NPOS_ - 1; if (id < 0) id = 0;
    idx[m] = id;
    if (mask1[br]) cp[m] = 0.f;
}

__global__ void k_count_r(const float* __restrict__ cp, int* __restrict__ r2gn) {
    int br = blockIdx.x * blockDim.x + threadIdx.x;
    if (br >= B_ * R_) return;
    int c = 0;
    for (int g = 0; g < G_; ++g) c += (cp[(ll)br * G_ + g] > 0.f);
    r2gn[br] = c;
}
__global__ void k_count_g(const float* __restrict__ cp, int* __restrict__ g2rn) {
    int bg = blockIdx.x * blockDim.x + threadIdx.x;
    if (bg >= B_ * G_) return;
    int b = bg / G_, g = bg % G_;
    int c = 0;
    for (int r = 0; r < R_; ++r) c += (cp[((ll)(b * R_ + r)) * G_ + g] > 0.f);
    g2rn[bg] = c;
}

__global__ void k_gather_extra(const int* __restrict__ idx, const float* __restrict__ sinus,
                               float* __restrict__ ea) {
    ll e = (ll)blockIdx.x * blockDim.x + threadIdx.x;
    if (e >= (ll)BRG_ * D_) return;
    int d = (int)(e & (D_ - 1)); ll m = e >> 9;
    ea[e] = sinus[(ll)idx[m] * D_ + d];
}

__global__ __launch_bounds__(256)
void k_att_sig(const float* __restrict__ regp, const float* __restrict__ gridp,
               float* __restrict__ attS) {
    int w = blockIdx.x * 8 + (threadIdx.x >> 5);
    if (w >= BRG_) return;
    int lane = threadIdx.x & 31;
    int g = w % G_; int br = w / G_; int b = br / R_;
    const float* x = regp  + (ll)br * D_;
    const float* y = gridp + (ll)(b * G_ + g) * D_;
    float s = 0.f;
    for (int k = lane; k < D_; k += 32) s += x[k] * y[k];
    #pragma unroll
    for (int off = 16; off; off >>= 1) s += __shfl_xor(s, off, 32);
    if (lane == 0) attS[w] = 1.f / (1.f + __expf(-s));
}

__global__ void k_att49(const float* __restrict__ attS, const float* __restrict__ Wm,
                        const float* __restrict__ bias, float* __restrict__ att) {
    int m = blockIdx.x * blockDim.x + threadIdx.x;
    if (m >= BRG_) return;
    int go = m % G_; int br = m / G_;
    float s = bias[go];
    for (int gi = 0; gi < G_; ++gi) s += attS[(ll)br * G_ + gi] * Wm[gi * G_ + go];
    att[m] = fmaxf(s, 0.f);
}
__global__ void k_att49_scale(float* __restrict__ att, const float* __restrict__ extra) {
    int br = blockIdx.x * blockDim.x + threadIdx.x;
    if (br >= B_ * R_) return;
    float am = -1e30f, em = -1e30f;
    for (int g = 0; g < G_; ++g) {
        am = fmaxf(am, att[(ll)br * G_ + g]);
        em = fmaxf(em, extra[(ll)br * G_ + g]);
    }
    float sc = em / (am * 3.f);
    for (int g = 0; g < G_; ++g)
        att[(ll)br * G_ + g] = att[(ll)br * G_ + g] * sc + extra[(ll)br * G_ + g];
}
__global__ void k_att30(const float* __restrict__ attS, const float* __restrict__ Wm,
                        const float* __restrict__ bias, float* __restrict__ att1) {
    int m = blockIdx.x * blockDim.x + threadIdx.x;
    if (m >= B_ * G_ * R_) return;
    int ro = m % R_; int bg = m / R_; int b = bg / G_; int g = bg % G_;
    float s = bias[ro];
    for (int ri = 0; ri < R_; ++ri) s += attS[((ll)(b * R_ + ri)) * G_ + g] * Wm[ri * R_ + ro];
    att1[m] = fmaxf(s, 0.f);
}
__global__ void k_att30_scale(float* __restrict__ att1, const float* __restrict__ extra) {
    int bg = blockIdx.x * blockDim.x + threadIdx.x;
    if (bg >= B_ * G_) return;
    int b = bg / G_, g = bg % G_;
    float am = -1e30f, em = -1e30f;
    for (int r = 0; r < R_; ++r) {
        am = fmaxf(am, att1[(ll)bg * R_ + r]);
        em = fmaxf(em, extra[((ll)(b * R_ + r)) * G_ + g]);
    }
    float sc = em / (am * 3.f);
    for (int r = 0; r < R_; ++r)
        att1[(ll)bg * R_ + r] = att1[(ll)bg * R_ + r] * sc + extra[((ll)(b * R_ + r)) * G_ + g];
}

__global__ void k_sort_fmask(const float* __restrict__ att, const int* __restrict__ cnt,
                             float* __restrict__ fm, int NQ, int NK) {
    int row = blockIdx.x * blockDim.x + threadIdx.x;
    if (row >= B_ * NQ) return;
    const float* a = att + (ll)row * NK;
    float* f = fm + (ll)row * NK;
    int nsel = cnt[row]; if (nsel > NK) nsel = NK;
    unsigned long long used = 0ull;
    for (int j = 0; j < NK; ++j) f[j] = 0.05f;
    for (int j = 0; j < nsel; ++j) {
        int best = -1; float bv = -1e30f;
        for (int g = 0; g < NK; ++g) {
            if (used & (1ull << g)) continue;
            float v = a[g];
            if (best < 0 || v >= bv) { bv = v; best = g; }  // ties -> larger index
        }
        used |= 1ull << best;
        f[best] = 1.0f;
    }
}

__global__ __launch_bounds__(256)
void k_fa_dot(const float* __restrict__ big, const float* __restrict__ ea,
              const float* __restrict__ Wf, float* __restrict__ fa, int Mrows) {
    int row = blockIdx.x * 8 + (threadIdx.x >> 5);
    if (row >= Mrows) return;
    int lane = threadIdx.x & 31;
    const float* a = big + (ll)row * D_;
    const float* e = ea  + (ll)row * D_;
    float s = 0.f;
    for (int k = lane; k < D_; k += 32) s += (a[k] + e[k]) * Wf[k];
    #pragma unroll
    for (int off = 16; off; off >>= 1) s += __shfl_xor(s, off, 32);
    if (lane == 0) fa[row] = fmaxf(s, 0.f);
}

__global__ void k_softmax_fa(float* __restrict__ fa, const float* __restrict__ fm,
                             int NQ, int NK) {
    int row = blockIdx.x * blockDim.x + threadIdx.x;
    if (row >= B_ * NQ) return;
    float* p = fa + (ll)row * NK;
    const float* m = fm + (ll)row * NK;
    float tmp[G_];
    float mx = -1e30f;
    for (int j = 0; j < NK; ++j) { float v = p[j] * m[j]; tmp[j] = v; mx = fmaxf(mx, v); }
    float s = 0.f;
    for (int j = 0; j < NK; ++j) { float e = __expf(tmp[j] - mx); tmp[j] = e; s += e; }
    float inv = 1.f / s;
    for (int j = 0; j < NK; ++j) p[j] = tmp[j] * inv;
}

__global__ __launch_bounds__(256)
void k_fused(const float* __restrict__ fa, const float* __restrict__ big,
             const float* __restrict__ keyf, float* __restrict__ out, int NQ, int NK) {
    int row = blockIdx.x;       // b*NQ + q
    int b = row / NQ;
    for (int d = threadIdx.x; d < D_; d += blockDim.x) {
        float acc = 0.f;
        for (int k = 0; k < NK; ++k) {
            float w = fa[(ll)row * NK + k];
            acc += w * big[((ll)row * NK + k) * D_ + d] * keyf[((ll)(b * NK + k)) * D_ + d];
        }
        out[(ll)row * D_ + d] = acc;
    }
}

__global__ void k_concat(const float* __restrict__ A, const float* __restrict__ Bm,
                         float* __restrict__ out, int M) {
    ll e = (ll)blockIdx.x * blockDim.x + threadIdx.x;
    if (e >= (ll)M * 2 * D_) return;
    int c = (int)(e & 1023); ll m = e >> 10;
    out[e] = (c < D_) ? A[m * D_ + c] : Bm[m * D_ + c - D_];
}

__global__ void k_out_image(const float* __restrict__ r2g, const float* __restrict__ g2r,
                            float* __restrict__ out) {
    ll e = (ll)blockIdx.x * blockDim.x + threadIdx.x;
    if (e >= (ll)B_ * (G_ + R_) * D_) return;
    int d = (int)(e & (D_ - 1)); ll bp = e >> 9;
    int pos = (int)(bp % (G_ + R_)); int b = (int)(bp / (G_ + R_));
    out[e] = (pos < G_) ? r2g[((ll)(b * G_ + pos)) * D_ + d]
                        : g2r[((ll)(b * R_ + (pos - G_))) * D_ + d];
}
__global__ void k_out_mask(const unsigned char* __restrict__ mask1, float* __restrict__ out) {
    int e = blockIdx.x * blockDim.x + threadIdx.x;
    if (e >= B_ * (G_ + R_)) return;
    int pos = e % (G_ + R_), b = e / (G_ + R_);
    out[e] = (pos < G_) ? 0.f : (mask1[b * R_ + (pos - G_)] ? 1.f : 0.f);
}
__global__ void k_copyf(const float* __restrict__ src, float* __restrict__ dst, ll n) {
    ll e = (ll)blockIdx.x * blockDim.x + threadIdx.x;
    if (e < n) dst[e] = src[e];
}
__global__ void k_out_m2(const unsigned char* __restrict__ mask0, float* __restrict__ out) {
    int e = blockIdx.x * blockDim.x + threadIdx.x;
    if (e < B_ * R_) out[e] = mask0[e] ? 1.f : 0.f;
}

// ============================ host side =====================================
static inline void launch_gemm(int amode, int act,
                               const float* A1, const float* A2,
                               int O1, int O2, ll s0, ll s1, ll s2,
                               const float* Wraw, const float* bias, float* Y,
                               int M, int N, int K, _Float16* wt, hipStream_t st)
{
    dim3 gt(K / 32, N / 32), bt(256);
    k_prep_w<<<gt, bt, 0, st>>>(Wraw, wt, K, N);
    dim3 g((M + 63) / 64, (N + 127) / 128), b(256);
    if (amode == 0) {
        if (act == 0)      gemm_wmma<0,0><<<g,b,0,st>>>(A1,A2,O1,O2,s0,s1,s2,wt,bias,Y,M,N,K);
        else if (act == 1) gemm_wmma<0,1><<<g,b,0,st>>>(A1,A2,O1,O2,s0,s1,s2,wt,bias,Y,M,N,K);
        else               gemm_wmma<0,2><<<g,b,0,st>>>(A1,A2,O1,O2,s0,s1,s2,wt,bias,Y,M,N,K);
    } else {
        if (act == 0)      gemm_wmma<1,0><<<g,b,0,st>>>(A1,A2,O1,O2,s0,s1,s2,wt,bias,Y,M,N,K);
        else if (act == 1) gemm_wmma<1,1><<<g,b,0,st>>>(A1,A2,O1,O2,s0,s1,s2,wt,bias,Y,M,N,K);
        else               gemm_wmma<1,2><<<g,b,0,st>>>(A1,A2,O1,O2,s0,s1,s2,wt,bias,Y,M,N,K);
    }
}

extern "C" void kernel_launch(void* const* d_in, const int* in_sizes, int n_in,
                              void* d_out, int out_size, void* d_ws, size_t ws_size,
                              hipStream_t stream)
{
    (void)in_sizes; (void)n_in; (void)out_size; (void)ws_size;
    #define PF(i) ((const float*)d_in[i])
    const float* grid_in  = PF(0);
    const int*   e2a      = (const int*)d_in[2];
    const int*   s2o      = (const int*)d_in[3];
    const unsigned char* sgm = (const unsigned char*)d_in[4];
    const unsigned char* mask0 = sgm;
    const unsigned char* mask1 = sgm + B_ * R_;
    const float* region_in = PF(5);
    const float* boxes_in  = PF(6);

    float* W = (float*)d_ws;
    size_t off = 0;
    auto F = [&](size_t n) { size_t o = off; off += (n + 63) & ~(size_t)63; return o; };
    const size_t oSIN   = F((size_t)NPOS_ * D_);
    const size_t oSG    = F((size_t)B_ * R_ * D_);
    const size_t oEABUF = F((size_t)B_ * R_ * 2 * D_);
    const size_t oRO    = F((size_t)B_ * R_ * D_);
    const size_t oMSG   = F((size_t)B_ * R_ * D_);
    const size_t oQ     = F((size_t)B_ * G_ * D_);
    const size_t oKb    = F((size_t)B_ * G_ * D_);
    const size_t oV     = F((size_t)B_ * G_ * D_);
    const size_t oO     = F((size_t)B_ * G_ * D_);
    const size_t oX     = F((size_t)B_ * G_ * D_);
    const size_t oHh    = F((size_t)B_ * G_ * DFF_);
    const size_t oWT    = F((size_t)D_ * DFF_ / 2 + 64);   // f16 scratch (N,K), max 512x2048
    const size_t oREGP  = F((size_t)B_ * R_ * D_);
    const size_t oGRIDP = F((size_t)B_ * G_ * D_);
    const size_t oG2R   = F((size_t)B_ * R_ * D_);
    const size_t oR2G   = F((size_t)B_ * G_ * D_);
    const size_t oCA    = F(BRG_);
    const size_t oCL    = F(BRG_);
    const size_t oCP    = F(BRG_);
    const size_t oEXTRA = F(BRG_);
    const size_t oMAX   = F(2);
    const size_t oIDX   = F(BRG_);          // int
    const size_t oR2GN  = F(B_ * R_);       // int
    const size_t oG2RN  = F(B_ * G_);       // int
    const size_t oATTS  = F(BRG_);
    const size_t oA49   = F(BRG_);
    const size_t oA30   = F(BRG_);
    const size_t oFM0   = F(BRG_);
    const size_t oFM1   = F(BRG_);
    const size_t oFA    = F(BRG_);
    const size_t oEA    = F((size_t)BRG_ * D_);
    const size_t oBIG   = F((size_t)BRG_ * D_);
    const size_t oTMP   = F((size_t)BRG_ * D_);
    const size_t oFUS0  = F((size_t)B_ * R_ * D_);
    const size_t oFUS1  = F((size_t)B_ * G_ * D_);
    const size_t oCAT   = F((size_t)B_ * G_ * 2 * D_);

    _Float16* WT = (_Float16*)(W + oWT);
    const int MR = B_ * R_;      // 960
    const int MG = B_ * G_;      // 1568
    const int MB = BRG_;         // 47040
    auto blks = [](ll n, int t) { return (int)((n + t - 1) / t); };

    k_zero2<<<1, 1, 0, stream>>>((unsigned*)(W + oMAX));
    k_sinusoid<<<blks(NPOS_ * D_, 256), 256, 0, stream>>>(W + oSIN);

    // ---------------- GCN + sg branch ----------------
    k_gather_embed2<<<blks((ll)MR * 2 * D_, 256), 256, 0, stream>>>(e2a, PF(45), W + oEABUF);
    launch_gemm(0, 1, W + oEABUF, nullptr, 1, 1, 2 * D_, 0, 0, PF(41), PF(43), W + oSG, MR, D_, 2 * D_, WT, stream);
    k_gather_relobj<<<blks((ll)MR * D_, 256), 256, 0, stream>>>(s2o, PF(45), W + oSG, W + oRO);
    launch_gemm(0, 1, W + oRO, nullptr, 1, 1, D_, 0, 0, PF(42), PF(44), W + oMSG, MR, D_, D_, WT, stream);
    k_scatter_add<<<blks((ll)MR * D_, 256), 256, 0, stream>>>(s2o, W + oMSG, W + oSG);

    auto run_mha = [&](float* X, int M, int ntok, int pb, const unsigned char* msk) {
        launch_gemm(0, 0, X, nullptr, 1, 1, D_, 0, 0, PF(pb + 2), PF(pb + 6), W + oQ,  M, D_, D_, WT, stream);
        launch_gemm(0, 0, X, nullptr, 1, 1, D_, 0, 0, PF(pb + 0), PF(pb + 4), W + oKb, M, D_, D_, WT, stream);
        launch_gemm(0, 0, X, nullptr, 1, 1, D_, 0, 0, PF(pb + 3), PF(pb + 7), W + oV,  M, D_, D_, WT, stream);
        k_attn<<<B_ * H_, 256, 0, stream>>>(W + oQ, W + oKb, W + oV, W + oO, msk, ntok);
        launch_gemm(0, 0, W + oO, nullptr, 1, 1, D_, 0, 0, PF(pb + 1), PF(pb + 5), W + oX, M, D_, D_, WT, stream);
        k_residual_ln<<<(M + 7) / 8, 256, 0, stream>>>(W + oX, X, nullptr, nullptr, X, M);
    };
    auto run_ffn = [&](float* X, int M, int pb) {
        launch_gemm(0, 1, X, nullptr, 1, 1, D_, 0, 0, PF(pb + 0), PF(pb + 2), W + oHh, M, DFF_, D_, WT, stream);
        launch_gemm(0, 0, W + oHh, nullptr, 1, 1, DFF_, 0, 0, PF(pb + 1), PF(pb + 3), W + oX, M, D_, DFF_, WT, stream);
        k_residual_ln<<<(M + 7) / 8, 256, 0, stream>>>(W + oX, X, nullptr, nullptr, X, M);
    };

    run_mha(W + oSG, MR, R_, 66, mask0);   // mha3
    run_ffn(W + oSG, MR, 23);              // ff3

    // ---------------- geometry ----------------
    k_geom1<<<blks(MB, 256), 256, 0, stream>>>(boxes_in, W + oCA, W + oCL, W + oCP, (unsigned*)(W + oMAX));
    k_geom2<<<blks(MB, 256), 256, 0, stream>>>(W + oCA, W + oCL, W + oCP, (unsigned*)(W + oMAX),
                                               mask1, W + oEXTRA, (int*)(W + oIDX));
    k_count_r<<<blks(MR, 256), 256, 0, stream>>>(W + oCP, (int*)(W + oR2GN));
    k_count_g<<<blks(MG, 256), 256, 0, stream>>>(W + oCP, (int*)(W + oG2RN));
    k_gather_extra<<<blks((ll)MB * D_, 256), 256, 0, stream>>>((int*)(W + oIDX), W + oSIN, W + oEA);

    // ---------------- projections + attention maps ----------------
    launch_gemm(0, 1, region_in, nullptr, 1, 1, D_, 0, 0, PF(10), PF(14), W + oREGP, MR, D_, D_, WT, stream);
    launch_gemm(0, 1, grid_in,   nullptr, 1, 1, D_, 0, 0, PF(9),  PF(13), W + oGRIDP, MG, D_, D_, WT, stream);
    k_att_sig<<<blks(MB, 256) , 256, 0, stream>>>(W + oREGP, W + oGRIDP, W + oATTS);
    k_att49<<<blks(MB, 256), 256, 0, stream>>>(W + oATTS, PF(8), PF(12), W + oA49);
    k_att49_scale<<<blks(MR, 256), 256, 0, stream>>>(W + oA49, W + oEXTRA);
    k_att30<<<blks(MB, 256), 256, 0, stream>>>(W + oATTS, PF(7), PF(11), W + oA30);
    k_att30_scale<<<blks(MG, 256), 256, 0, stream>>>(W + oA30, W + oEXTRA);
    k_sort_fmask<<<blks(MR, 256), 256, 0, stream>>>(W + oA49, (int*)(W + oR2GN), W + oFM0, R_, G_);
    k_sort_fmask<<<blks(MG, 256), 256, 0, stream>>>(W + oA30, (int*)(W + oG2RN), W + oFM1, G_, R_);

    // ---------------- fusion 0 (region queries grid) ----------------
    launch_gemm(1, 2, W + oREGP, W + oGRIDP, R_, G_, 0, 0, 0, PF(27), PF(31), W + oBIG, MB, D_, D_, WT, stream);
    launch_gemm(0, 1, W + oEA, nullptr, 1, 1, D_, 0, 0, PF(28), PF(32), W + oTMP, MB, D_, D_, WT, stream);
    k_fa_dot<<<blks(MB, 256), 256, 0, stream>>>(W + oBIG, W + oTMP, PF(29), W + oFA, MB);
    k_softmax_fa<<<blks(MR, 256), 256, 0, stream>>>(W + oFA, W + oFM0, R_, G_);
    k_fused<<<MR, 256, 0, stream>>>(W + oFA, W + oBIG, W + oGRIDP, W + oFUS0, R_, G_);
    k_concat<<<blks((ll)MR * 2 * D_, 256), 256, 0, stream>>>(W + oREGP, W + oFUS0, W + oCAT, MR);
    launch_gemm(0, 1, W + oCAT, nullptr, 1, 1, 2 * D_, 0, 0, PF(30), PF(33), W + oX, MR, D_, 2 * D_, WT, stream);
    k_residual_ln<<<(MR + 7) / 8, 256, 0, stream>>>(W + oX, region_in, PF(49), PF(48), W + oG2R, MR);
    run_mha(W + oG2R, MR, R_, 58, mask1);  // mha2
    run_ffn(W + oG2R, MR, 19);             // ff2

    // ---------------- fusion 1 (grid queries region) ----------------
    launch_gemm(1, 2, W + oGRIDP, W + oREGP, G_, R_, 0, 0, 0, PF(34), PF(38), W + oBIG, MB, D_, D_, WT, stream);
    launch_gemm(0, 1, W + oEA, nullptr, G_, R_, (ll)R_ * G_ * D_, D_, (ll)G_ * D_,
                PF(35), PF(39), W + oTMP, MB, D_, D_, WT, stream);          // transposed read
    k_fa_dot<<<blks(MB, 256), 256, 0, stream>>>(W + oBIG, W + oTMP, PF(36), W + oFA, MB);
    k_softmax_fa<<<blks(MG, 256), 256, 0, stream>>>(W + oFA, W + oFM1, G_, R_);
    k_fused<<<MG, 256, 0, stream>>>(W + oFA, W + oBIG, W + oREGP, W + oFUS1, G_, R_);
    k_concat<<<blks((ll)MG * 2 * D_, 256), 256, 0, stream>>>(W + oGRIDP, W + oFUS1, W + oCAT, MG);
    launch_gemm(0, 1, W + oCAT, nullptr, 1, 1, 2 * D_, 0, 0, PF(37), PF(40), W + oX, MG, D_, 2 * D_, WT, stream);
    k_residual_ln<<<(MG + 7) / 8, 256, 0, stream>>>(W + oX, grid_in, PF(47), PF(46), W + oR2G, MG);
    run_mha(W + oR2G, MG, G_, 50, nullptr); // mha1
    run_ffn(W + oR2G, MG, 15);              // ff1

    // ---------------- outputs ----------------
    float* out = (float*)d_out;
    const ll oImg = 0;
    const ll oMsk = (ll)B_ * (G_ + R_) * D_;
    const ll oSgO = oMsk + (ll)B_ * (G_ + R_);
    const ll oM2  = oSgO + (ll)B_ * R_ * D_;
    k_out_image<<<blks((ll)B_ * (G_ + R_) * D_, 256), 256, 0, stream>>>(W + oR2G, W + oG2R, out + oImg);
    k_out_mask<<<blks(B_ * (G_ + R_), 256), 256, 0, stream>>>(mask1, out + oMsk);
    k_copyf<<<blks((ll)B_ * R_ * D_, 256), 256, 0, stream>>>(W + oSG, out + oSgO, (ll)B_ * R_ * D_);
    k_out_m2<<<blks(B_ * R_, 256), 256, 0, stream>>>(mask0, out + oM2);
    #undef PF
}
